// Linear_27324581937317
// MI455X (gfx1250) — 8-bit matrix operand order repaired and checked by reading, compiled, not run on hardware
//
#include <hip/hip_runtime.h>
#include <stdint.h>

#define M_DIM 4096
#define N_DIM 4096
#define K_DIM 4096

typedef __attribute__((ext_vector_type(8))) int v8i;

__global__ void pack_x_kernel(const int* __restrict__ x, const int* __restrict__ x_zp,
                              uint32_t* __restrict__ out, int n4) {
    int i = blockIdx.x * blockDim.x + threadIdx.x;
    if (i >= n4) return;
    const int zp = x_zp[0];
    int4 v = ((const int4*)x)[i];
    uint32_t b0 = (uint32_t)((v.x - zp) & 0xff);
    uint32_t b1 = (uint32_t)((v.y - zp) & 0xff);
    uint32_t b2 = (uint32_t)((v.z - zp) & 0xff);
    uint32_t b3 = (uint32_t)((v.w - zp) & 0xff);
    out[i] = b0 | (b1 << 8) | (b2 << 16) | (b3 << 24);
}

__global__ void pack_w_kernel(const int* __restrict__ w, uint32_t* __restrict__ out, int n4) {
    int i = blockIdx.x * blockDim.x + threadIdx.x;
    if (i >= n4) return;
    int4 v = ((const int4*)w)[i];
    uint32_t b0 = (uint32_t)(v.x & 0xff);
    uint32_t b1 = (uint32_t)(v.y & 0xff);
    uint32_t b2 = (uint32_t)(v.z & 0xff);
    uint32_t b3 = (uint32_t)(v.w & 0xff);
    out[i] = b0 | (b1 << 8) | (b2 << 16) | (b3 << 24);
}

__global__ void __launch_bounds__(256)
gemm_i8_wmma_kernel(const int8_t* __restrict__ A8,
                    const int8_t* __restrict__ W8,
                    const float*  __restrict__ bias,
                    const float*  __restrict__ x_scale,
                    const float*  __restrict__ w_scale,
                    const float*  __restrict__ out_scale,
                    const int*    __restrict__ out_zp,
                    float*        __restrict__ out) {
    const int lane  = threadIdx.x;
    const int wave  = threadIdx.y;
    const int lmod  = lane & 15;
    const int lhalf = lane >> 4;

    const int n0 = (blockIdx.x * 8 + wave) * 64;
    const int m0 = blockIdx.y * 64;

    const int8_t* aBase = A8 + (size_t)(m0 + lmod) * K_DIM + lhalf * 16;
    const int8_t* bBase = W8 + (size_t)(n0 + lmod) * K_DIM + lhalf * 16;

    v8i acc[4][4] = {};

    for (int k = 0; k < K_DIM; k += 64) {
        if (k + 64 < K_DIM) {
            __builtin_prefetch(aBase + k + 64, 0, 1);
            __builtin_prefetch(bBase + k + 64, 0, 1);
        }

        v8i b[4];
#pragma unroll
        for (int j = 0; j < 4; ++j) {
            const int8_t* p = bBase + (size_t)(j * 16) * K_DIM + k;
            int4 q0 = *(const int4*)(p +  0);
            int4 q1 = *(const int4*)(p + 32);
            v8i f;
            f[0] = q0.x; f[1] = q0.y; f[2] = q0.z; f[3] = q0.w;
            f[4] = q1.x; f[5] = q1.y; f[6] = q1.z; f[7] = q1.w;
            b[j] = f;
        }

        v8i a[4];
#pragma unroll
        for (int i = 0; i < 4; ++i) {
            const int8_t* p = aBase + (size_t)(i * 16) * K_DIM + k;
            int2 t0 = *(const int2*)(p +  0);
            int2 t1 = *(const int2*)(p +  8);
            int2 t2 = *(const int2*)(p + 32);
            int2 t3 = *(const int2*)(p + 40);
            v8i f;
            f[0] = t0.x; f[1] = t0.y;
            f[2] = t1.x; f[3] = t1.y;
            f[4] = t2.x; f[5] = t2.y;
            f[6] = t3.x; f[7] = t3.y;
            a[i] = f;
        }

#pragma unroll
        for (int i = 0; i < 4; ++i)
#pragma unroll
            for (int j = 0; j < 4; ++j)
                acc[i][j] = __builtin_amdgcn_wmma_i32_16x16x64_iu8(
                     true, a[i],  true, b[j],
                    acc[i][j],  false,  false);
    }

    const float s      = x_scale[0] * w_scale[0];
    const float os     = out_scale[0];
    const int   zp_out = out_zp[0];

#pragma unroll
    for (int j = 0; j < 4; ++j) {
        const int   col  = n0 + j * 16 + lmod;
        const float bcol = bias[col];
#pragma unroll
        for (int i = 0; i < 4; ++i) {
            const int rowBase = m0 + i * 16 + 8 * lhalf;
#pragma unroll
            for (int r = 0; r < 8; ++r) {
                float y = (float)acc[i][j][r] * s + bcol;
                int   q = (int)rintf(y / os) + zp_out;
                q = q < 0 ? 0 : (q > 255 ? 255 : q);
                out[(size_t)(rowBase + r) * N_DIM + col] = (float)q;
            }
        }
    }
}

extern "C" void kernel_launch(void* const* d_in, const int* in_sizes, int n_in,
                              void* d_out, int out_size, void* d_ws, size_t ws_size,
                              hipStream_t stream) {
    const int*   x_q       = (const int*)d_in[0];
    const int*   w_q       = (const int*)d_in[1];
    const float* bias      = (const float*)d_in[2];
    const float* x_scale   = (const float*)d_in[3];
    const int*   x_zp      = (const int*)d_in[4];
    const float* w_scale   = (const float*)d_in[5];
    const float* out_scale = (const float*)d_in[6];
    const int*   out_zp    = (const int*)d_in[7];

    int8_t* A8 = (int8_t*)d_ws;
    int8_t* W8 = A8 + (size_t)M_DIM * K_DIM;

    {
        const int n4 = (M_DIM * K_DIM) / 4;
        pack_x_kernel<<<(n4 + 255) / 256, 256, 0, stream>>>(x_q, x_zp, (uint32_t*)A8, n4);
    }
    {
        const int n4 = (N_DIM * K_DIM) / 4;
        pack_w_kernel<<<(n4 + 255) / 256, 256, 0, stream>>>(w_q, (uint32_t*)W8, n4);
    }

    dim3 block(32, 8);
    dim3 grid(N_DIM / (64 * 8), M_DIM / 64);
    gemm_i8_wmma_kernel<<<grid, block, 0, stream>>>(
        A8, W8, bias, x_scale, w_scale, out_scale, out_zp, (float*)d_out);
}
